// EdgeFluxModule_49357764165711
// MI455X (gfx1250) — matrix operand order repaired and checked by reading, not compiled, not run on hardware
//
#include <hip/hip_runtime.h>
#include <hip/hip_bf16.h>

// ---------------------------------------------------------------------------
// EdgeFlux module for MI455X (gfx1250, wave32, WMMA f16 16x16x32).
//
// N=50000, E=500000, D=64, MSG_IN=292 (K-padded to 320).
// ~61 GFLOP total, ~450 MB HBM traffic. The binding resource is WGP$/L2
// weight re-fetch for WMMA B-fragments, so each wave processes TWO 16-edge
// M-tiles and reuses every B-fragment across both (halves weight traffic).
// edge_memory rows are prefetched global->LDS via the CDNA5 async path
// (ASYNCcnt) and consumed by the LayerNorm stage.
//
// d_in flatten order (setup_inputs dict order, params alphabetical,
// MLP tuples (W1,b1,W2,b2)):
//  0 node_u[N*5] 1 edge_index[2E,i64] 2 edge_attr[E*3] 3 mean[2] 4 std[2]
//  5 edge_memory[E*64]
//  6 input_proj.W 7 input_proj.b 8 ln_beta 9 ln_gamma
// 10 phi_edge.W1 11 .b1 12 .W2 13 .b2
// 14 phi_msg.W1 15 .b1 16 .W2 17 .b2
// 18 phi_node.W1 19 .b1 20 .W2 21 .b2
// 22..25 psi_e  26..29 psi_rho  30..33 psi_rhou  34..37 psi_visc
//
// d_out: F_rho[0,2E) F_e[2E,4E) F_rhou[4E,6E) n[6E,8E) r[8E,9E)
//        alpha[9E,10E) m_ij[10E,74E)
// ---------------------------------------------------------------------------

typedef __attribute__((ext_vector_type(16))) _Float16 v16h;
typedef __attribute__((ext_vector_type(8)))  _Float16 v8h;
typedef __attribute__((ext_vector_type(8)))  float    v8f;
typedef __attribute__((ext_vector_type(4)))  float    v4f;
typedef __attribute__((ext_vector_type(4)))  int      v4i;

#define WMMA_F16(a, b, c) \
  __builtin_amdgcn_wmma_f32_16x16x32_f16(false, (a), false, (b), (short)0, (c), false, false)

#if __has_builtin(__builtin_amdgcn_global_load_async_to_lds_b128)
#define ASYNC_LDS 1
#else
#define ASYNC_LDS 0
#endif

__device__ __forceinline__ float gelu_f(float x) {
  return 0.5f * x * (1.0f + erff(x * 0.70710678118654752f));
}

// Wave-level LDS producer->consumer fence (single-wave workgroups).
__device__ __forceinline__ void wave_lds_sync() {
  __builtin_amdgcn_wave_barrier();
  asm volatile("s_wait_dscnt 0" ::: "memory");
  __builtin_amdgcn_wave_barrier();
}

__device__ __forceinline__ void async_copy_b128(const void* g, void* l) {
#if ASYNC_LDS
  // param0: global source (pointer to 16B int vector), param1: LDS dest.
  __builtin_amdgcn_global_load_async_to_lds_b128((v4i*)g, (v4i*)l, 0, 0);
#else
  *(v4f*)l = *(const v4f*)g;
#endif
}

__device__ __forceinline__ void async_wait0() {
#if ASYNC_LDS
#if __has_builtin(__builtin_amdgcn_s_wait_asynccnt)
  __builtin_amdgcn_s_wait_asynccnt(0);
#else
  asm volatile("s_wait_asynccnt 0x0" ::: "memory");
#endif
#endif
}

// A-fragment (16x32 f16, row m = lane&15) from LDS row-major [16][ldk] halves.
// ISA 7.12.2: halves 0..7 -> K = kc + half*8 + i, 8..15 -> K = kc+16+half*8+i.
__device__ __forceinline__ v16h load_a_frag(const _Float16* M, int m, int half,
                                            int kc, int ldk) {
  const v8h a0 = *(const v8h*)(M + m * ldk + kc + half * 8);
  const v8h a1 = *(const v8h*)(M + m * ldk + kc + 16 + half * 8);
  return __builtin_shufflevector(a0, a1, 0, 1, 2, 3, 4, 5, 6, 7,
                                 8, 9, 10, 11, 12, 13, 14, 15);
}

// B-fragment (32x16 f16, col n = lane&15) from global W[out][ldk] f16 rows:
// halves 0..7 -> K = kc + half*8 + i, 8..15 -> K = kc+16+half*8+i.
__device__ __forceinline__ v16h load_b_frag(const _Float16* W, int n, int half,
                                            int nb, int kc, int ldk) {
  const _Float16* p = W + (long long)(nb + n) * (long long)ldk + kc + half * 8;
  const v8h b0 = *(const v8h*)(p);
  const v8h b1 = *(const v8h*)(p + 16);
  return __builtin_shufflevector(b0, b1, 0, 1, 2, 3, 4, 5, 6, 7,
                                 8, 9, 10, 11, 12, 13, 14, 15);
}

// ---------------------------------------------------------------------------
// Weight prep: f32 -> f16 WMMA-friendly layouts in workspace.
// ---------------------------------------------------------------------------
__global__ void prep_weights_kernel(
    const float* __restrict__ msgW1, const float* __restrict__ msgW2,
    const float* __restrict__ rhoW1, const float* __restrict__ eW1,
    const float* __restrict__ rhouW1, const float* __restrict__ viscW1,
    const float* __restrict__ rhob1, const float* __restrict__ eb1,
    const float* __restrict__ rhoub1, const float* __restrict__ viscb1,
    _Float16* __restrict__ Whm1, _Float16* __restrict__ Whm2,
    _Float16* __restrict__ Wcat, float* __restrict__ bcat) {
  int t = blockIdx.x * blockDim.x + threadIdx.x;
  if (t < 128 * 320) {
    int o = t / 320, k = t % 320;
    Whm1[t] = (k < 292) ? (_Float16)msgW1[k * 128 + o] : (_Float16)0.0f;
    return;
  }
  t -= 128 * 320;
  if (t < 64 * 128) {
    int o = t / 128, k = t % 128;
    Whm2[t] = (_Float16)msgW2[k * 64 + o];
    return;
  }
  t -= 64 * 128;
  if (t < 224 * 64) {
    int o = t / 64, k = t % 64;
    float v;
    if (o < 64)       v = rhoW1[k * 64 + o];
    else if (o < 128) v = eW1[k * 64 + (o - 64)];
    else if (o < 192) v = rhouW1[k * 64 + (o - 128)];
    else              v = viscW1[k * 32 + (o - 192)];
    Wcat[t] = (_Float16)v;
    return;
  }
  t -= 224 * 64;
  if (t < 224) {
    float b;
    if (t < 64)       b = rhob1[t];
    else if (t < 128) b = eb1[t - 64];
    else if (t < 192) b = rhoub1[t - 128];
    else              b = viscb1[t - 192];
    bcat[t] = b;
  }
}

// ---------------------------------------------------------------------------
// Node kernel: h = MLP_5_64_64(u) + u @ Wp + bp  (0.45 GFLOP; scalar).
// ---------------------------------------------------------------------------
__global__ void __launch_bounds__(128) node_kernel(
    const float* __restrict__ node_u,
    const float* __restrict__ nW1, const float* __restrict__ nb1,
    const float* __restrict__ nW2, const float* __restrict__ nb2,
    const float* __restrict__ pW, const float* __restrict__ pb,
    _Float16* __restrict__ hbuf, int N) {
  int i = blockIdx.x * blockDim.x + threadIdx.x;
  if (i >= N) return;
  const float u0 = node_u[i * 5 + 0], u1 = node_u[i * 5 + 1],
              u2 = node_u[i * 5 + 2], u3 = node_u[i * 5 + 3],
              u4 = node_u[i * 5 + 4];
  float hid[64];
#pragma unroll
  for (int j = 0; j < 64; ++j) {
    float a = nb1[j] + u0 * nW1[j] + u1 * nW1[64 + j] + u2 * nW1[128 + j] +
              u3 * nW1[192 + j] + u4 * nW1[256 + j];
    hid[j] = gelu_f(a);
  }
  for (int o = 0; o < 64; ++o) {
    float a = nb2[o] + pb[o] + u0 * pW[o] + u1 * pW[64 + o] + u2 * pW[128 + o] +
              u3 * pW[192 + o] + u4 * pW[256 + o];
#pragma unroll
    for (int j = 0; j < 64; ++j) a += hid[j] * nW2[j * 64 + o];
    hbuf[(long long)i * 64 + o] = (_Float16)a;
  }
}

// ---------------------------------------------------------------------------
// Edge kernel: ONE wave (32 threads) per block; each wave owns 32 edges as
// two 16-row M-tiles so every WMMA B-fragment is reused for 2 WMMAs.
// Per-wave LDS (49,152 B, overlaid):
//   [0,20480)       msgh    f16 [2][16][320]  (dead after GEMM1)
//   [20480,28672)   hiddenh f16 [2][16][128]  (dead after GEMM2)
//   [0,28672)       catt    f32 [2][16][224]  (overlay, written after GEMM3)
//   [28672,36864)   mtile   f32 [2][16][64]
//   [36864,40960)   mijh    f16 [2][16][64]
//   [40960,49152)   emem    f32 [32][64]      (async-prefetched edge_memory)
// ---------------------------------------------------------------------------
__global__ void __launch_bounds__(32) edge_kernel(
    const float* __restrict__ node_u, const long long* __restrict__ eidx,
    const float* __restrict__ edge_attr, const float* __restrict__ mean_mom,
    const float* __restrict__ std_mom, const float* __restrict__ edge_mem,
    const float* __restrict__ phiE_W1, const float* __restrict__ phiE_b1,
    const float* __restrict__ phiE_W2, const float* __restrict__ phiE_b2,
    const float* __restrict__ msg_b1, const float* __restrict__ msg_b2,
    const float* __restrict__ ln_gamma, const float* __restrict__ ln_beta,
    const float* __restrict__ rhoW2, const float* __restrict__ rhob2,
    const float* __restrict__ eW2, const float* __restrict__ eb2,
    const float* __restrict__ rhouW2, const float* __restrict__ rhoub2,
    const float* __restrict__ viscW2, const float* __restrict__ viscb2,
    const _Float16* __restrict__ hbuf, const _Float16* __restrict__ Whm1,
    const _Float16* __restrict__ Whm2, const _Float16* __restrict__ Wcat,
    const float* __restrict__ bcat, float* __restrict__ out, int E) {
  __shared__ __align__(16) unsigned char smem[49152];

  const int lane = threadIdx.x;   // 0..31, owns edge e0+lane in scalar stages
  const int j  = lane & 15;       // WMMA row/col within a 16-subtile
  const int hi = lane >> 4;       // WMMA half-wave id == owned M-tile index

  const long long e0 = (long long)blockIdx.x * 32;
  if (e0 >= E) return;  // wave-uniform; EXEC stays all-ones for WMMA

  _Float16* msgh    = (_Float16*)smem;             // [2][16][320]
  _Float16* hiddenh = (_Float16*)(smem + 20480);   // [2][16][128]
  float*    catt    = (float*)smem;                // [2][16][224] overlay
  float*    mtile   = (float*)(smem + 28672);      // [2][16][64]
  _Float16* mijh    = (_Float16*)(smem + 36864);   // [2][16][64]
  float*    emem    = (float*)(smem + 40960);      // [32][64]

  const int e = (int)e0 + lane;
  const bool valid = (e < E);
  const int ec = valid ? e : (E - 1);  // clamp loads; stores are guarded

  // ---- async prefetch: edge_memory row -> LDS (consumed at LayerNorm) -----
  {
    const float* emrow = edge_mem + (long long)ec * 64;
#pragma unroll
    for (int k = 0; k < 64; k += 4)
      async_copy_b128(emrow + k, &emem[lane * 64 + k]);
  }

  // ---- stage 1: geometry, gathers, phi_edge, msg_in assembly --------------
  const long long src = eidx[ec];
  const long long dst = eidx[(long long)E + ec];
  const float dx = edge_attr[ec * 3 + 0];
  const float dy = edge_attr[ec * 3 + 1];
  const float rr = edge_attr[ec * 3 + 2];
  const float inv = 1.0f / (rr + 1e-12f);
  const float nx = dx * inv, ny = dy * inv;
  const float tx = -ny, ty = nx;

  _Float16* msgrow = msgh + lane * 320;  // tile hi, row j

  // gather h_src/h_dst; v_avg & v_diff computed in registers on the fly
  {
    const _Float16* hs = hbuf + src * 64;
    const _Float16* hd = hbuf + dst * 64;
#pragma unroll
    for (int k = 0; k < 64; k += 8) {
      const v8h s = *(const v8h*)(hs + k);
      const v8h d = *(const v8h*)(hd + k);
      v8h av, df;
#pragma unroll
      for (int i = 0; i < 8; ++i) {
        const float fs = (float)s[i], fd = (float)d[i];
        av[i] = (_Float16)(fs + fd);
        const float dv = fs - fd;
        df[i] = (_Float16)(dv * dv);
      }
      *(v8h*)(msgrow + k) = s;
      *(v8h*)(msgrow + 64 + k) = d;
      *(v8h*)(msgrow + 128 + k) = av;
      *(v8h*)(msgrow + 192 + k) = df;
    }
  }

  // phi_edge: 1->32->32 MLP on r
  {
    float he[32];
#pragma unroll
    for (int k = 0; k < 32; ++k) he[k] = gelu_f(rr * phiE_W1[k] + phiE_b1[k]);
    for (int o = 0; o < 32; ++o) {
      float a = phiE_b2[o];
#pragma unroll
      for (int k = 0; k < 32; ++k) a += he[k] * phiE_W2[k * 32 + o];
      msgrow[256 + o] = (_Float16)a;
    }
  }

  // vel_proj (4 entries) + K padding
  {
    const float s0 = std_mom[0], s1 = std_mom[1];
    const float m0 = mean_mom[0], m1 = mean_mom[1];
    const float stdn = sqrtf(s0 * s0 + s1 * s1) + 1e-8f;
    const float pxs = node_u[src * 5 + 3] * s0 + m0;
    const float pys = node_u[src * 5 + 4] * s1 + m1;
    const float pxd = node_u[dst * 5 + 3] * s0 + m0;
    const float pyd = node_u[dst * 5 + 4] * s1 + m1;
    msgrow[288] = (_Float16)((pxs * nx + pys * ny) / stdn);
    msgrow[289] = (_Float16)((pxs * tx + pys * ty) / stdn);
    msgrow[290] = (_Float16)((pxd * nx + pyd * ny) / stdn);
    msgrow[291] = (_Float16)((pxd * tx + pyd * ty) / stdn);
#pragma unroll
    for (int k = 292; k < 320; ++k) msgrow[k] = (_Float16)0.0f;
  }

  // state diffs + n/r outputs
  const float du0 = node_u[dst * 5 + 0] - node_u[src * 5 + 0];
  const float du1 = node_u[dst * 5 + 1] - node_u[src * 5 + 1];
  const float du3 = node_u[dst * 5 + 3] - node_u[src * 5 + 3];
  const float du4 = node_u[dst * 5 + 4] - node_u[src * 5 + 4];
  if (valid) {
    out[6LL * E + 2LL * e + 0] = nx;
    out[6LL * E + 2LL * e + 1] = ny;
    out[8LL * E + e] = rr;
  }
  wave_lds_sync();

  // ---- GEMM1: hidden(2x16x128) = msg(2x16x320) @ Whm1^T, gelu -------------
  // N-loop split in two passes of 4 subtiles; each B-frag feeds 2 WMMAs.
  for (int nbh = 0; nbh < 8; nbh += 4) {
    v8f acc0[4], acc1[4];
    const v8f z = {};
#pragma unroll
    for (int q = 0; q < 4; ++q) { acc0[q] = z; acc1[q] = z; }
    for (int kc = 0; kc < 320; kc += 32) {
      const v16h a0 = load_a_frag(msgh,        j, hi, kc, 320);
      const v16h a1 = load_a_frag(msgh + 5120, j, hi, kc, 320);
#pragma unroll
      for (int q = 0; q < 4; ++q) {
        const v16h b = load_b_frag(Whm1, j, hi, (nbh + q) * 16, kc, 320);
        acc0[q] = WMMA_F16(a0, b, acc0[q]);
        acc1[q] = WMMA_F16(a1, b, acc1[q]);
      }
    }
#pragma unroll
    for (int q = 0; q < 4; ++q) {
      const int o = (nbh + q) * 16 + j;
      const float bb = msg_b1[o];
#pragma unroll
      for (int r = 0; r < 8; ++r) {
        hiddenh[(r + 8 * hi) * 128 + o]        = (_Float16)gelu_f(acc0[q][r] + bb);
        hiddenh[2048 + (r + 8 * hi) * 128 + o] = (_Float16)gelu_f(acc1[q][r] + bb);
      }
    }
  }
  wave_lds_sync();

  // ---- GEMM2: m(2x16x64) = hidden(2x16x128) @ Whm2^T + b2 -----------------
  {
    v8f acc0[4], acc1[4];
    const v8f z = {};
#pragma unroll
    for (int q = 0; q < 4; ++q) { acc0[q] = z; acc1[q] = z; }
    for (int kc = 0; kc < 128; kc += 32) {
      const v16h a0 = load_a_frag(hiddenh,        j, hi, kc, 128);
      const v16h a1 = load_a_frag(hiddenh + 2048, j, hi, kc, 128);
#pragma unroll
      for (int q = 0; q < 4; ++q) {
        const v16h b = load_b_frag(Whm2, j, hi, q * 16, kc, 128);
        acc0[q] = WMMA_F16(a0, b, acc0[q]);
        acc1[q] = WMMA_F16(a1, b, acc1[q]);
      }
    }
#pragma unroll
    for (int q = 0; q < 4; ++q) {
      const int o = q * 16 + j;
      const float bb = msg_b2[o];
#pragma unroll
      for (int r = 0; r < 8; ++r) {
        mtile[(r + 8 * hi) * 64 + o]        = acc0[q][r] + bb;
        mtile[1024 + (r + 8 * hi) * 64 + o] = acc1[q][r] + bb;
      }
    }
  }
  async_wait0();   // edge_memory rows resident in LDS
  wave_lds_sync();

  // ---- LayerNorm(m + edge_memory) -> m_ij (out + f16 tile), all 32 lanes --
  {
    float* mrow = mtile + hi * 1024 + j * 64;
    const float* emrow = emem + lane * 64;
    float mu = 0.f;
    for (int k = 0; k < 64; ++k) {
      const float x = mrow[k] + emrow[k];
      mrow[k] = x;
      mu += x;
    }
    mu *= (1.0f / 64.0f);
    float var = 0.f;
    for (int k = 0; k < 64; ++k) {
      const float d = mrow[k] - mu;
      var += d * d;
    }
    var *= (1.0f / 64.0f);
    const float rs = rsqrtf(var + 1e-5f);
    _Float16* mijrow = mijh + hi * 1024 + j * 64;
    for (int k = 0; k < 64; ++k) {
      const float mij = (mrow[k] - mu) * rs * ln_gamma[k] + ln_beta[k];
      if (valid) out[10LL * E + (long long)e * 64 + k] = mij;
      mijrow[k] = (_Float16)mij;
    }
  }
  wave_lds_sync();

  // ---- GEMM3: fused psi layer-1: cat(2x16x224) = m_ij @ Wcat^T, gelu ------
  for (int nb = 0; nb < 14; ++nb) {
    v8f c0 = {}, c1 = {};
#pragma unroll
    for (int kc = 0; kc < 64; kc += 32) {
      const v16h a0 = load_a_frag(mijh,        j, hi, kc, 64);
      const v16h a1 = load_a_frag(mijh + 1024, j, hi, kc, 64);
      const v16h b = load_b_frag(Wcat, j, hi, nb * 16, kc, 64);
      c0 = WMMA_F16(a0, b, c0);
      c1 = WMMA_F16(a1, b, c1);
    }
    const int o = nb * 16 + j;
    const float bb = bcat[o];
#pragma unroll
    for (int r = 0; r < 8; ++r) {
      catt[(r + 8 * hi) * 224 + o]        = gelu_f(c0[r] + bb);
      catt[3584 + (r + 8 * hi) * 224 + o] = gelu_f(c1[r] + bb);
    }
  }
  wave_lds_sync();

  // ---- psi layer-2 heads + flux assembly (all 32 lanes, ~290 MACs/edge) ---
  if (valid) {
    float arho = rhob2[0], ae = eb2[0];
    float ar0 = rhoub2[0], ar1 = rhoub2[1];
    float vv = viscb2[0];
    const float* crow = catt + hi * 3584 + j * 224;
#pragma unroll 8
    for (int h = 0; h < 64; ++h) {
      arho += crow[h] * rhoW2[h];
      ae   += crow[64 + h] * eW2[h];
      ar0  += crow[128 + h] * rhouW2[h * 2 + 0];
      ar1  += crow[128 + h] * rhouW2[h * 2 + 1];
    }
#pragma unroll 8
    for (int h = 0; h < 32; ++h) vv += crow[192 + h] * viscW2[h];
    const float alpha = 1.0f / (1.0f + expf(-vv));

    out[2LL * e + 0] = arho * nx - alpha * du0 * nx;
    out[2LL * e + 1] = arho * ny - alpha * du0 * ny;
    out[2LL * E + 2LL * e + 0] = ae * nx - alpha * du1 * nx;
    out[2LL * E + 2LL * e + 1] = ae * ny - alpha * du1 * ny;
    out[4LL * E + 2LL * e + 0] = ar0 * nx + ar1 * tx - alpha * du3;
    out[4LL * E + 2LL * e + 1] = ar0 * ny + ar1 * ty - alpha * du4;
    out[9LL * E + e] = alpha;
  }
}

// ---------------------------------------------------------------------------
extern "C" void kernel_launch(void* const* d_in, const int* in_sizes, int n_in,
                              void* d_out, int out_size, void* d_ws,
                              size_t ws_size, hipStream_t stream) {
  const int N = 50000;
  const int E = 500000;

  const float*     node_u    = (const float*)d_in[0];
  const long long* eidx      = (const long long*)d_in[1];
  const float*     edge_attr = (const float*)d_in[2];
  const float*     mean_mom  = (const float*)d_in[3];
  const float*     std_mom   = (const float*)d_in[4];
  const float*     edge_mem  = (const float*)d_in[5];
  const float*     ipW       = (const float*)d_in[6];
  const float*     ipb       = (const float*)d_in[7];
  const float*     ln_beta   = (const float*)d_in[8];
  const float*     ln_gamma  = (const float*)d_in[9];
  const float*     phiE_W1   = (const float*)d_in[10];
  const float*     phiE_b1   = (const float*)d_in[11];
  const float*     phiE_W2   = (const float*)d_in[12];
  const float*     phiE_b2   = (const float*)d_in[13];
  const float*     msgW1     = (const float*)d_in[14];
  const float*     msgb1     = (const float*)d_in[15];
  const float*     msgW2     = (const float*)d_in[16];
  const float*     msgb2     = (const float*)d_in[17];
  const float*     nW1       = (const float*)d_in[18];
  const float*     nb1       = (const float*)d_in[19];
  const float*     nW2       = (const float*)d_in[20];
  const float*     nb2       = (const float*)d_in[21];
  const float*     eW1       = (const float*)d_in[22];
  const float*     eb1       = (const float*)d_in[23];
  const float*     eW2p      = (const float*)d_in[24];
  const float*     eb2       = (const float*)d_in[25];
  const float*     rhoW1     = (const float*)d_in[26];
  const float*     rhob1     = (const float*)d_in[27];
  const float*     rhoW2     = (const float*)d_in[28];
  const float*     rhob2     = (const float*)d_in[29];
  const float*     rhouW1    = (const float*)d_in[30];
  const float*     rhoub1    = (const float*)d_in[31];
  const float*     rhouW2    = (const float*)d_in[32];
  const float*     rhoub2    = (const float*)d_in[33];
  const float*     viscW1    = (const float*)d_in[34];
  const float*     viscb1    = (const float*)d_in[35];
  const float*     viscW2    = (const float*)d_in[36];
  const float*     viscb2    = (const float*)d_in[37];

  // workspace layout (256B-aligned offsets): requires ~6.53 MB
  char* ws = (char*)d_ws;
  _Float16* hbuf = (_Float16*)ws;                    // N*64 f16 = 6,400,000 B
  _Float16* Whm1 = (_Float16*)(ws + 6400000);        // 128*320 f16 = 81,920 B
  _Float16* Whm2 = (_Float16*)(ws + 6481920);        // 64*128 f16 = 16,384 B
  _Float16* Wcat = (_Float16*)(ws + 6498304);        // 224*64 f16 = 28,672 B
  float*    bcat = (float*)(ws + 6526976);           // 224 f32

  const int prep_total = 128 * 320 + 64 * 128 + 224 * 64 + 224;
  prep_weights_kernel<<<(prep_total + 255) / 256, 256, 0, stream>>>(
      msgW1, msgW2, rhoW1, eW1, rhouW1, viscW1, rhob1, eb1, rhoub1, viscb1,
      Whm1, Whm2, Wcat, bcat);

  node_kernel<<<(N + 127) / 128, 128, 0, stream>>>(node_u, nW1, nb1, nW2, nb2,
                                                   ipW, ipb, hbuf, N);

  edge_kernel<<<(E + 31) / 32, 32, 0, stream>>>(
      node_u, eidx, edge_attr, mean_mom, std_mom, edge_mem, phiE_W1, phiE_b1,
      phiE_W2, phiE_b2, msgb1, msgb2, ln_gamma, ln_beta, rhoW2, rhob2, eW2p,
      eb2, rhouW2, rhoub2, viscW2, viscb2, hbuf, Whm1, Whm2, Wcat, bcat,
      (float*)d_out, E);
}
